// GNNLeakDetector_12266426597591
// MI455X (gfx1250) — matrix operand order repaired and checked by reading, not compiled, not run on hardware
//
#include <hip/hip_runtime.h>
#include <hip/hip_bf16.h>
#include <math.h>

// ---------------------------------------------------------------------------
// GNN leak detector (2x NNConv + scatter-mean + sigmoid head) for gfx1250.
// Dominant compute (E x 4160 x 64 bilinear GEMM of conv2) runs on
// v_wmma_f32_16x16x32_f16 with on-the-fly A construction (r (x) h_src outer
// product rows) and an L2-resident pre-transposed f16 B (W2b + b2b folded in).
// ---------------------------------------------------------------------------

typedef _Float16 v16h __attribute__((ext_vector_type(16)));
typedef _Float16 v8h  __attribute__((ext_vector_type(8)));
typedef float    v8f  __attribute__((ext_vector_type(8)));

#define HID   64
#define KTOT  4160      // 4096 (k*64+i) + 64 (bias block, virtual r = 1)
#define TILE_E 32       // edges per block in the WMMA kernel (2 subtiles of 16)

// ---------------------------------------------------------------- zero fill
__global__ void k_zero_f32(float* __restrict__ p, long n) {
    long t = (long)blockIdx.x * blockDim.x + threadIdx.x;
    if (t < n) p[t] = 0.0f;
}

// ------------------------------------------------- build Bt[n][K] f16 (L2)
// Bt[o][k*64+i] = W2b[k*4096 + i*64 + o]   (k,i,o in 0..63)
// Bt[o][4096+i] = b2b [i*64 + o]
__global__ void k_prep_bt(const float* __restrict__ W2b,
                          const float* __restrict__ b2b,
                          _Float16* __restrict__ bt) {
    int t = blockIdx.x * blockDim.x + threadIdx.x;
    if (t >= HID * KTOT) return;
    int n  = t / KTOT;
    int kk = t - n * KTOT;
    float v;
    if (kk < 4096) {
        int k = kk >> 6, i = kk & 63;
        v = W2b[k * 4096 + i * 64 + n];
    } else {
        v = b2b[(kk - 4096) * 64 + n];
    }
    bt[(long)n * KTOT + kk] = (_Float16)v;
}

// --------------------------------------------------- conv1 edge pass (f32)
// 256 threads = 4 edges x 64 output channels.
__global__ void __launch_bounds__(256)
k_conv1_edge(const float* __restrict__ x, const int* __restrict__ ei,
             const float* __restrict__ ea,
             const float* __restrict__ W1a, const float* __restrict__ b1a,
             const float* __restrict__ W2a, const float* __restrict__ b2a,
             float* __restrict__ agg1, float* __restrict__ cnt, int E) {
    __shared__ float sz[4][HID];
    __shared__ float sxs[4][4];
    __shared__ int   sdst[4];

    const int g = threadIdx.x >> 6;   // local edge 0..3
    const int o = threadIdx.x & 63;   // output channel
    const int e = blockIdx.x * 4 + g;

    if (e < E) {
        if (o == 0) sdst[g] = ei[E + e];
        if (o < 4) { int s = ei[e]; sxs[g][o] = x[s * 4 + o]; }
        float a0 = ea[e * 3 + 0], a1 = ea[e * 3 + 1], a2 = ea[e * 3 + 2];
        float z = b1a[o] + a0 * W1a[o] + a1 * W1a[64 + o] + a2 * W1a[128 + o];
        sz[g][o] = fmaxf(z, 0.0f);
    }
    __syncthreads();
    if (e < E) {
        float x0 = sxs[g][0], x1 = sxs[g][1], x2 = sxs[g][2], x3 = sxs[g][3];
        float acc = x0 * b2a[o] + x1 * b2a[64 + o] + x2 * b2a[128 + o] + x3 * b2a[192 + o];
#pragma unroll 4
        for (int k = 0; k < HID; ++k) {
            const float* w = W2a + k * 256 + o;
            float wk = x0 * w[0] + x1 * w[64] + x2 * w[128] + x3 * w[192];
            acc = fmaf(sz[g][k], wk, acc);
        }
        atomicAdd(&agg1[(long)sdst[g] * HID + o], acc);
        if (o == 0) atomicAdd(&cnt[sdst[g]], 1.0f);
    }
}

// -------------------------------------- node update 1: h = relu(mean + root)
__global__ void k_node1(const float* __restrict__ x, const float* __restrict__ agg1,
                        const float* __restrict__ cnt, const float* __restrict__ root1,
                        const float* __restrict__ bias1, float* __restrict__ h, int N) {
    int t = blockIdx.x * blockDim.x + threadIdx.x;
    if (t >= N * HID) return;
    int n = t >> 6, o = t & 63;
    float inv = 1.0f / fmaxf(cnt[n], 1.0f);
    const float* xr = x + (long)n * 4;
    float v = agg1[t] * inv + bias1[o]
            + xr[0] * root1[o] + xr[1] * root1[64 + o]
            + xr[2] * root1[128 + o] + xr[3] * root1[192 + o];
    h[t] = fmaxf(v, 0.0f);
}

// ------------------------------------------------- conv2 edge pass (WMMA)
// Per block: 32 edges, 4 waves (128 threads). Wave w owns output cols
// [16w, 16w+16). 130 K-steps of v_wmma_f32_16x16x32_f16 per subtile.
// Branch-free hot loop: h slices hoisted to registers, r read in v8h chunks,
// bias block (virtual r = 1) handled in a multiply-free epilogue.
__global__ void __launch_bounds__(128)
k_conv2_wmma(const float* __restrict__ h, const int* __restrict__ ei,
             const float* __restrict__ ea,
             const float* __restrict__ W1b, const float* __restrict__ b1b,
             const _Float16* __restrict__ bt,
             float* __restrict__ agg2, int E) {
    __shared__ __align__(32) _Float16 sh_h[TILE_E][72];  // padded rows (144B)
    __shared__ __align__(32) _Float16 sh_r[TILE_E][72];  // r, row-major by edge
    __shared__ int   s_src[TILE_E];
    __shared__ int   s_dst[TILE_E];

    const int tid = threadIdx.x;
    const int e0  = blockIdx.x * TILE_E;

    if (tid < TILE_E) {
        int ge = e0 + tid;
        if (ge < E) { s_src[tid] = ei[ge]; s_dst[tid] = ei[E + ge]; }
        else        { s_src[tid] = 0;      s_dst[tid] = -1;         }
    }
    __syncthreads();

    // r = relu(ea@W1b + b1b) and gather h_src, both as f16
    for (int idx = tid; idx < TILE_E * HID; idx += 128) {
        int e = idx >> 6, o = idx & 63;
        int ge = e0 + e;
        if (ge < E) {
            float a0 = ea[ge * 3 + 0], a1 = ea[ge * 3 + 1], a2 = ea[ge * 3 + 2];
            float z = b1b[o] + a0 * W1b[o] + a1 * W1b[64 + o] + a2 * W1b[128 + o];
            sh_r[e][o] = (_Float16)fmaxf(z, 0.0f);
            sh_h[e][o] = (_Float16)h[(long)s_src[e] * HID + o];
        } else {
            sh_r[e][o] = (_Float16)0.0f;
            sh_h[e][o] = (_Float16)0.0f;
        }
    }
    __syncthreads();

    const int wave = tid >> 5;
    const int lane = tid & 31;
    const int m    = lane & 15;      // A row / D column-within-tile selector
    const int hi   = lane >> 4;      // which lane-half
    const int ncol = wave * 16 + m;  // output column via B fragment
    const int q0   = hi ? 16 : 0;    // A-fragment half offsets (contiguous k = 16*hi + s)
    const int q1   = hi ? 24 : 8;

    // Hoist the lane's h slices: 4 v8h per subtile row, reused by all k-steps.
    const _Float16* hr0 = &sh_h[m][0];
    const _Float16* hr1 = &sh_h[16 + m][0];
    const v8h h0a = *(const v8h*)(hr0 + q0);
    const v8h h0b = *(const v8h*)(hr0 + q1);
    const v8h h0c = *(const v8h*)(hr0 + 32 + q0);
    const v8h h0d = *(const v8h*)(hr0 + 32 + q1);
    const v8h h1a = *(const v8h*)(hr1 + q0);
    const v8h h1b = *(const v8h*)(hr1 + q1);
    const v8h h1c = *(const v8h*)(hr1 + 32 + q0);
    const v8h h1d = *(const v8h*)(hr1 + 32 + q1);

    v8f acc0 = {}; v8f acc1 = {};
    const _Float16* btrow = bt + (long)ncol * KTOT + hi * 16;

    // Main loop: kidx chunks of 8 (one v8h of r per subtile per chunk).
    for (int cc = 0; cc < 8; ++cc) {
        const v8h r0 = *(const v8h*)(&sh_r[m][cc * 8]);
        const v8h r1 = *(const v8h*)(&sh_r[16 + m][cc * 8]);
#pragma unroll
        for (int j = 0; j < 8; ++j) {
#pragma unroll
            for (int half = 0; half < 2; ++half) {
                const int kb = (cc * 8 + j) * 2 + half;
                const v16h b = *(const v16h*)(btrow + kb * 32);

                const v8h pa = half ? h0c : h0a;
                const v8h pb = half ? h0d : h0b;
                const _Float16 rv0 = r0[j];
                union { v16h v; _Float16 e[16]; } a0;
#pragma unroll
                for (int t = 0; t < 8; ++t) { a0.e[t] = pa[t] * rv0; a0.e[8 + t] = pb[t] * rv0; }
                acc0 = __builtin_amdgcn_wmma_f32_16x16x32_f16(
                    false, a0.v, false, b, (short)0, acc0, false, false);

                const v8h qa = half ? h1c : h1a;
                const v8h qb = half ? h1d : h1b;
                const _Float16 rv1 = r1[j];
                union { v16h v; _Float16 e[16]; } a1;
#pragma unroll
                for (int t = 0; t < 8; ++t) { a1.e[t] = qa[t] * rv1; a1.e[8 + t] = qb[t] * rv1; }
                acc1 = __builtin_amdgcn_wmma_f32_16x16x32_f16(
                    false, a1.v, false, b, (short)0, acc1, false, false);
            }
        }
    }

    // Bias epilogue (K = 4096..4159): virtual r = 1, A = h slices directly.
#pragma unroll
    for (int half = 0; half < 2; ++half) {
        const int kb = 128 + half;
        const v16h b = *(const v16h*)(btrow + kb * 32);

        const v8h pa = half ? h0c : h0a;
        const v8h pb = half ? h0d : h0b;
        union { v16h v; _Float16 e[16]; } a0;
#pragma unroll
        for (int t = 0; t < 8; ++t) { a0.e[t] = pa[t]; a0.e[8 + t] = pb[t]; }
        acc0 = __builtin_amdgcn_wmma_f32_16x16x32_f16(
            false, a0.v, false, b, (short)0, acc0, false, false);

        const v8h qa = half ? h1c : h1a;
        const v8h qb = half ? h1d : h1b;
        union { v16h v; _Float16 e[16]; } a1;
#pragma unroll
        for (int t = 0; t < 8; ++t) { a1.e[t] = qa[t]; a1.e[8 + t] = qb[t]; }
        acc1 = __builtin_amdgcn_wmma_f32_16x16x32_f16(
            false, a1.v, false, b, (short)0, acc1, false, false);
    }

    // Scatter: D layout -> VGPR v holds edge (v + 8*hi), column ncol.
#pragma unroll
    for (int v = 0; v < 8; ++v) {
        int mm = v + hi * 8;
        int d0 = s_dst[mm];
        int d1 = s_dst[16 + mm];
        if (d0 >= 0) atomicAdd(&agg2[(long)d0 * HID + ncol], acc0[v]);
        if (d1 >= 0) atomicAdd(&agg2[(long)d1 * HID + ncol], acc1[v]);
    }
}

// ------------------- node update 2 + output head: sigmoid(relu(...) @ Wout)
__global__ void __launch_bounds__(256)
k_node2_out(const float* __restrict__ h, const float* __restrict__ agg2,
            const float* __restrict__ cnt, const float* __restrict__ root2,
            const float* __restrict__ bias2, const float* __restrict__ Wout,
            const float* __restrict__ bout, float* __restrict__ out, int N) {
    __shared__ float sval[4][HID];
    const int g = threadIdx.x >> 6;
    const int o = threadIdx.x & 63;
    const int n = blockIdx.x * 4 + g;

    float v = 0.0f;
    if (n < N) {
        float inv = 1.0f / fmaxf(cnt[n], 1.0f);
        v = agg2[(long)n * HID + o] * inv + bias2[o];
        const float* hn = h + (long)n * HID;
#pragma unroll 4
        for (int k = 0; k < HID; ++k) v = fmaf(hn[k], root2[k * 64 + o], v);
        v = fmaxf(v, 0.0f) * Wout[o];
    }
    sval[g][o] = v;
    __syncthreads();
#pragma unroll
    for (int s = 32; s > 0; s >>= 1) {
        if (o < s) sval[g][o] += sval[g][o + s];
        __syncthreads();
    }
    if (o == 0 && n < N) {
        float t = sval[g][0] + bout[0];
        out[n] = 1.0f / (1.0f + expf(-t));
    }
}

// ---------------------------------------------------------------------------
extern "C" void kernel_launch(void* const* d_in, const int* in_sizes, int n_in,
                              void* d_out, int out_size, void* d_ws, size_t ws_size,
                              hipStream_t stream) {
    const float* x     = (const float*)d_in[0];
    const int*   ei    = (const int*)  d_in[1];
    const float* ea    = (const float*)d_in[2];
    const float* W1a   = (const float*)d_in[3];
    const float* b1a   = (const float*)d_in[4];
    const float* W2a   = (const float*)d_in[5];
    const float* b2a   = (const float*)d_in[6];
    const float* root1 = (const float*)d_in[7];
    const float* bias1 = (const float*)d_in[8];
    const float* W1b   = (const float*)d_in[9];
    const float* b1b   = (const float*)d_in[10];
    const float* W2b   = (const float*)d_in[11];
    const float* b2b   = (const float*)d_in[12];
    const float* root2 = (const float*)d_in[13];
    const float* bias2 = (const float*)d_in[14];
    const float* Wout  = (const float*)d_in[15];
    const float* bout  = (const float*)d_in[16];
    float* out = (float*)d_out;

    const int N = in_sizes[0] / 4;   // 40000
    const int E = in_sizes[1] / 2;   // 100000

    // Workspace carve-out (256B aligned): agg1, agg2, cnt | h | Bt
    char* wsb = (char*)d_ws;
    size_t off = 0;
    auto carve = [&](size_t bytes) -> char* {
        char* p = wsb + off;
        off = (off + bytes + 255) & ~(size_t)255;
        return p;
    };
    float*    agg1 = (float*)   carve((size_t)N * HID * sizeof(float));
    float*    agg2 = (float*)   carve((size_t)N * HID * sizeof(float));
    float*    cnt  = (float*)   carve((size_t)N * sizeof(float));
    size_t zero_bytes = off;                 // zero everything up to here
    float*    hbuf = (float*)   carve((size_t)N * HID * sizeof(float));
    _Float16* bt   = (_Float16*)carve((size_t)HID * KTOT * sizeof(_Float16));
    (void)ws_size; (void)n_in; (void)out_size;

    // 1) zero accumulators + counts (ws is poisoned 0xAA by the harness)
    {
        long nz = (long)(zero_bytes / sizeof(float));
        int blocks = (int)((nz + 255) / 256);
        k_zero_f32<<<blocks, 256, 0, stream>>>((float*)wsb, nz);
    }
    // 2) build f16 transposed B for the conv2 WMMA GEMM (W2b + b2b folded)
    {
        int total = HID * KTOT;
        k_prep_bt<<<(total + 255) / 256, 256, 0, stream>>>(W2b, b2b, bt);
    }
    // 3) conv1 edge messages + per-dst counts
    k_conv1_edge<<<(E + 3) / 4, 256, 0, stream>>>(x, ei, ea, W1a, b1a, W2a, b2a,
                                                  agg1, cnt, E);
    // 4) node update 1 -> h
    k_node1<<<(N * HID + 255) / 256, 256, 0, stream>>>(x, agg1, cnt, root1, bias1,
                                                       hbuf, N);
    // 5) conv2 edge messages via WMMA bilinear GEMM
    k_conv2_wmma<<<(E + TILE_E - 1) / TILE_E, 128, 0, stream>>>(hbuf, ei, ea, W1b,
                                                                b1b, bt, agg2, E);
    // 6) node update 2 + sigmoid head
    k_node2_out<<<(N + 3) / 4, 256, 0, stream>>>(hbuf, agg2, cnt, root2, bias2,
                                                 Wout, bout, out, N);
}
